// FeatNeighbourCorr_85993835201175
// MI455X (gfx1250) — compile-verified, matrix operand order checked by reading, not run on hardware
//
#include <hip/hip_runtime.h>
#include <hip/hip_bf16.h>

typedef _Float16 v16h __attribute__((ext_vector_type(16)));
typedef _Float16 v8h  __attribute__((ext_vector_type(8)));
typedef float    v8f  __attribute__((ext_vector_type(8)));

#define C_DIM 128
#define H_DIM 256
#define W_DIM 256
#define TILE  16
#define HALO  18                    // TILE + 2 (reflect pad d=1)
#define CPAD  136                   // 128 + 8 halves -> pixel stride 272B = 68 dwords (bank stride 4)
#define NPIX  (HALO * HALO)         // 324
#define SMEM_HALVES (NPIX * CPAD)   // 44064 halves = 88128 B
#define SMEM_BYTES  (SMEM_HALVES * 2 + NPIX * 4)  // + inv-norm array = 89424 B (< 320KB WGP LDS)

__device__ __forceinline__ int refl(int i, int n) {
  if (i < 0)  i = -i;            // reflect (no edge repeat), d==1 so single fold suffices
  if (i >= n) i = 2 * n - 2 - i;
  return i;
}

__launch_bounds__(256)
__global__ void corr8_wmma_kernel(const float* __restrict__ feats,
                                  float* __restrict__ out) {
  extern __shared__ _Float16 smem[];          // [NPIX][CPAD] raw f16 features
  float* inv = (float*)(smem + SMEM_HALVES);  // [NPIX] 1/||f||

  const int tid = threadIdx.x;
  const int b   = blockIdx.z;
  const int h0  = blockIdx.y * TILE;
  const int w0  = blockIdx.x * TILE;

  // ---------- Phase 1: HBM -> LDS, fp32 -> f16, reflect-padded 18x18 halo tile ----------
  const float* fb = feats + (size_t)b * C_DIM * H_DIM * W_DIM;
  for (int i = tid; i < NPIX * C_DIM; i += 256) {
    int c = i / NPIX;
    int p = i - c * NPIX;         // p fastest with tid -> coalesced rows in W
    int y = p / HALO;
    int x = p - y * HALO;
    int hh = refl(h0 + y - 1, H_DIM);
    int ww = refl(w0 + x - 1, W_DIM);
    float v = fb[((size_t)c * H_DIM + hh) * W_DIM + ww];
    smem[p * CPAD + c] = (_Float16)v;
  }
  __syncthreads();

  // ---------- Phase 2: per-pixel inverse L2 norm ----------
  for (int p = tid; p < NPIX; p += 256) {
    const v8h* row = (const v8h*)(smem + p * CPAD);
    float s = 0.f;
    #pragma unroll
    for (int q = 0; q < C_DIM / 8; ++q) {
      v8h h8 = row[q];
      #pragma unroll
      for (int e = 0; e < 8; ++e) { float f = (float)h8[e]; s += f * f; }
    }
    inv[p] = rsqrtf(s);
  }
  __syncthreads();

  // ---------- Phase 3: Gram diagonals via v_wmma_f32_16x16x32_f16 ----------
  const int wv   = tid >> 5;                 // 8 waves
  const int lane = tid & 31;
  const int Mn   = lane & 15;                // A-row / B-col index this lane supplies
  const int koff = (lane < 16) ? 0 : 8;      // f16 A/B fragment K-offset per ISA 7.12.2
  const int sel  = lane & 7;
  const bool diag_valid = (lane < 8) || (lane >= 24);
  const int pix  = (lane < 8) ? lane : (lane - 16);

  const int DX[8] = {1, 1, 0, -1, -1, -1, 0, 1};   // neighbor = (h - dx, w - dy)
  const int DY[8] = {0, 1, 1,  1,  0, -1, -1, -1};

  for (int rr = 0; rr < 2; ++rr) {
    const int r = wv + rr * 8;               // tile row 0..15
    // Load A fragments (center pixels of row r) once; reuse across all 8 offsets.
    const _Float16* cen = smem + (size_t)((r + 1) * HALO + (Mn + 1)) * CPAD;
    v16h A[4];
    #pragma unroll
    for (int q = 0; q < 4; ++q) {
      int kb = q * 32 + koff;                // two 8-half runs: kb..kb+7 and kb+16..kb+23
      v8h lo = *(const v8h*)(cen + kb);      // ds_load_b128, 16B aligned, bank-conflict-free
      v8h hi = *(const v8h*)(cen + kb + 16);
      A[q] = __builtin_shufflevector(lo, hi, 0,1,2,3,4,5,6,7,8,9,10,11,12,13,14,15);
    }
    #pragma unroll
    for (int o = 0; o < 8; ++o) {
      const int dx = DX[o], dy = DY[o];
      const _Float16* nbr =
          smem + (size_t)((r + 1 - dx) * HALO + (Mn + 1 - dy)) * CPAD;
      v8f acc = {};
      #pragma unroll
      for (int q = 0; q < 4; ++q) {
        int kb = q * 32 + koff;
        v8h lo = *(const v8h*)(nbr + kb);
        v8h hi = *(const v8h*)(nbr + kb + 16);
        v16h Bf = __builtin_shufflevector(lo, hi, 0,1,2,3,4,5,6,7,8,9,10,11,12,13,14,15);
        // D += A x B : D[i][j] = dot(center_i, neighbor_j); diagonal j==i is the answer.
        acc = __builtin_amdgcn_wmma_f32_16x16x32_f16(
            /*neg_a=*/false, A[q], /*neg_b=*/false, Bf,
            /*c_mod=*/(short)0, acc, /*reuse_a=*/false, /*reuse_b=*/false);
      }
      // Extract D[i,i]: lanes 0-7 (vgpr=lane) and 24-31 (vgpr=lane-24) hold it.
      float t0 = (sel & 1) ? acc[1] : acc[0];
      float t1 = (sel & 1) ? acc[3] : acc[2];
      float t2 = (sel & 1) ? acc[5] : acc[4];
      float t3 = (sel & 1) ? acc[7] : acc[6];
      float u0 = (sel & 2) ? t1 : t0;
      float u1 = (sel & 2) ? t3 : t2;
      float d  = (sel & 4) ? u1 : u0;
      float res = d * inv[(r + 1) * HALO + (pix + 1)]
                    * inv[(r + 1 - dx) * HALO + (pix + 1 - dy)];
      if (diag_valid) {
        out[(((size_t)b * 8 + o) * H_DIM + (h0 + r)) * W_DIM + (w0 + pix)] = res;
      }
    }
  }
}

extern "C" void kernel_launch(void* const* d_in, const int* in_sizes, int n_in,
                              void* d_out, int out_size, void* d_ws, size_t ws_size,
                              hipStream_t stream) {
  const float* feats = (const float*)d_in[0];
  float* out = (float*)d_out;

  const int Bn = in_sizes[0] / (C_DIM * H_DIM * W_DIM);  // 8

  // Dynamic LDS > default 64KB cap: raise the limit (idempotent, capture-safe).
  (void)hipFuncSetAttribute((const void*)corr8_wmma_kernel,
                            hipFuncAttributeMaxDynamicSharedMemorySize, SMEM_BYTES);

  dim3 grid(W_DIM / TILE, H_DIM / TILE, Bn);
  dim3 block(256);
  corr8_wmma_kernel<<<grid, block, SMEM_BYTES, stream>>>(feats, out);
}
